// HebbianBlock_65231963291892
// MI455X (gfx1250) — matrix operand order repaired and checked by reading, not compiled, not run on hardware
//
#include <hip/hip_runtime.h>
#include <math.h>

typedef __bf16 bf16;
typedef __attribute__((ext_vector_type(16))) __bf16 v16bf;
typedef __attribute__((ext_vector_type(8)))  __bf16 v8bf;
typedef __attribute__((ext_vector_type(8)))  float  v8f;

constexpr int Bb = 2;
constexpr int T  = 8192;
constexpr int D  = 1024;
constexpr int H  = 8;
constexpr int HD = 128;    // head dim
constexpr int C  = 64;     // chunk size
constexpr int NC = T / C;  // 128 chunks
constexpr int BT = Bb * T;
constexpr int BH = Bb * H;
constexpr int G   = 4;       // dv-split groups in the scan
constexpr int DVL = HD / G;  // 32 dv columns per scan workgroup

// ---------------------------------------------------------------------------
// WMMA helpers (wave32, 16x16x32 bf16 -> f32)
// ---------------------------------------------------------------------------
__device__ __forceinline__ v8f wmma_bb(v16bf a, v16bf b, v8f c) {
  return __builtin_amdgcn_wmma_f32_16x16x32_bf16(false, a, false, b,
                                                 (short)0, c, false, false);
}

__device__ __forceinline__ v8f zero8() {
  v8f z;
#pragma unroll
  for (int i = 0; i < 8; ++i) z[i] = 0.f;
  return z;
}

// A fragment (16x32, MxK): lane (l=lane&15) holds row m0+l.
// K layout: lanes<16 -> K {0..7, 16..23}; lanes>=16 -> K {8..15, 24..31}.
__device__ __forceinline__ v16bf load_frag_a(const bf16* base, int ld) {
  int lane = threadIdx.x & 31;
  int l = lane & 15, hi = lane >> 4;
  const bf16* p = base + (size_t)l * ld + hi * 8;
  v8bf x0 = *(const v8bf*)(p);
  v8bf x1 = *(const v8bf*)(p + 16);
  v16bf a;
#pragma unroll
  for (int i = 0; i < 8; ++i) { a[i] = x0[i]; a[8 + i] = x1[i]; }
  return a;
}

__device__ __forceinline__ v16bf load_frag_a_scale(const bf16* base, int ld, float s) {
  v16bf a = load_frag_a(base, ld);
#pragma unroll
  for (int i = 0; i < 16; ++i) a[i] = (bf16)((float)a[i] * s);
  return a;
}

// B fragment (32x16, KxN) from [n][k] layout (k contiguous, stride ld):
// lane (l=lane&15) holds column n0+l; K layout same as A: lanes<16 -> K {0..7, 16..23}; lanes>=16 -> K {8..15, 24..31}.
__device__ __forceinline__ v16bf load_frag_b(const bf16* base, int ld) {
  int lane = threadIdx.x & 31;
  int l = lane & 15, hi = lane >> 4;
  const bf16* p = base + (size_t)l * ld + hi * 8;
  v8bf x0 = *(const v8bf*)(p);
  v8bf x1 = *(const v8bf*)(p + 16);
  v16bf b;
#pragma unroll
  for (int i = 0; i < 8; ++i) { b[i] = x0[i]; b[8 + i] = x1[i]; }
  return b;
}

// ---------------------------------------------------------------------------
// Kernel 0: weights -> bf16
// ---------------------------------------------------------------------------
__global__ void convw_kernel(const float* __restrict__ Ww, const float* __restrict__ Wr,
                             bf16* __restrict__ Wwb, bf16* __restrict__ Wrb) {
  int i = blockIdx.x * 256 + threadIdx.x;
  if (i < D * D) { Wwb[i] = (bf16)Ww[i]; Wrb[i] = (bf16)Wr[i]; }
}

// ---------------------------------------------------------------------------
// Kernel 1: per-token prep: normalized k (bf16), beta = sigmoid(x @ Wb^T),
// bf16 copy of x. One block per token; wave w handles head w.
// ---------------------------------------------------------------------------
__global__ __launch_bounds__(256) void prep_kernel(
    const float* __restrict__ x, const float* __restrict__ Wbeta,
    bf16* __restrict__ kn, bf16* __restrict__ xb, float* __restrict__ beta) {
  __shared__ float xr[D];
  int token = blockIdx.x;          // b*T + t
  int b = token >> 13, t = token & (T - 1);
  int tid = threadIdx.x;
#pragma unroll
  for (int i = 0; i < 4; ++i) {
    float v = x[(size_t)token * D + tid + 256 * i];
    xr[tid + 256 * i] = v;
    xb[(size_t)token * D + tid + 256 * i] = (bf16)v;
  }
  __syncthreads();
  int h = tid >> 5, s = tid & 31;
  float ss = 0.f;
#pragma unroll
  for (int j = 0; j < 4; ++j) { float v = xr[h * HD + s + 32 * j]; ss += v * v; }
  float bd = 0.f;
  for (int j = 0; j < 32; ++j) bd += xr[s + 32 * j] * Wbeta[(size_t)h * D + s + 32 * j];
#pragma unroll
  for (int o = 16; o > 0; o >>= 1) {
    ss += __shfl_xor(ss, o, 32);
    bd += __shfl_xor(bd, o, 32);
  }
  float rn = 1.f / fmaxf(sqrtf(ss), 1e-12f);
  if (s == 0) beta[((size_t)(b * H + h)) * T + t] = 1.f / (1.f + __expf(-bd));
  size_t kbase = (((size_t)(b * H + h)) * T + t) * HD;
#pragma unroll
  for (int j = 0; j < 4; ++j) {
    int dk = s + 32 * j;
    kn[kbase + dk] = (bf16)(xr[h * HD + dk] * rn);
  }
}

// ---------------------------------------------------------------------------
// Kernel 2: v = (x @ Ww^T) * beta, stored transposed per chunk:
// vT[bh][nch][dv][c]  (K-major for the next WMMA stage)
// ---------------------------------------------------------------------------
__global__ __launch_bounds__(256) void gemm_v_kernel(
    const bf16* __restrict__ xb, const bf16* __restrict__ Wwb,
    const float* __restrict__ beta, bf16* __restrict__ vT) {
  int wave = threadIdx.x >> 5, lane = threadIdx.x & 31;
  int l = lane & 15, hi = lane >> 4;
  int m0 = blockIdx.x * 128 + (wave >> 1) * 32;
  int n0 = blockIdx.y * 128 + (wave & 1) * 64;
  v8f acc[2][4];
#pragma unroll
  for (int i = 0; i < 2; ++i)
#pragma unroll
    for (int j = 0; j < 4; ++j) acc[i][j] = zero8();
  for (int k0 = 0; k0 < D; k0 += 32) {
    v16bf af[2], bfm[4];
#pragma unroll
    for (int i = 0; i < 2; ++i)
      af[i] = load_frag_a(xb + (size_t)(m0 + 16 * i) * D + k0, D);
#pragma unroll
    for (int j = 0; j < 4; ++j)
      bfm[j] = load_frag_b(Wwb + (size_t)(n0 + 16 * j) * D + k0, D);
#pragma unroll
    for (int i = 0; i < 2; ++i)
#pragma unroll
      for (int j = 0; j < 4; ++j) acc[i][j] = wmma_bb(af[i], bfm[j], acc[i][j]);
  }
#pragma unroll
  for (int i = 0; i < 2; ++i) {
#pragma unroll
    for (int j = 0; j < 4; ++j) {
      int n = n0 + 16 * j + l;
      int h = n >> 7, dv = n & 127;
#pragma unroll
      for (int r = 0; r < 8; ++r) {
        int m = m0 + 16 * i + hi * 8 + r;
        int b = m >> 13, t = m & (T - 1);
        float bet = beta[((size_t)(b * H + h)) * T + t];
        int nch = t >> 6, cc = t & 63;
        vT[(((size_t)((b * H + h) * NC + nch)) * HD + dv) * C + cc] =
            (bf16)(acc[i][j][r] * bet);
      }
    }
  }
}

// ---------------------------------------------------------------------------
// Kernel 3: per-chunk UT transform. One block per (b,h,chunk).
//   A0 = -tril(k_beta k^T * L, -1) ; A = (I - A0)^{-1} (forward subst)
//   vpr = A @ v ; kcd = A @ (k_beta * exp(cumdecay))
// ---------------------------------------------------------------------------
__global__ __launch_bounds__(256) void ut_kernel(
    const bf16* __restrict__ kn, const float* __restrict__ beta,
    const float* __restrict__ decay, const bf16* __restrict__ vT,
    float* __restrict__ vpr, bf16* __restrict__ kcd) {
  __shared__ float A0[C * C];
  __shared__ bf16  Ab[C * C];
  __shared__ bf16  kbdeT[HD * C];   // [dk][c]
  __shared__ float lexpTab[C];
  __shared__ float deTab[C];
  int blk = blockIdx.x;             // bh*NC + nch
  int bh = blk >> 7, nch = blk & (NC - 1);
  int h = bh & 7;
  int tid = threadIdx.x;
  int wave = tid >> 5, lane = tid & 31, l = lane & 15, hi = lane >> 4;
  float lg = logf(1.f / (1.f + __expf(-decay[h])));
  const bf16* kchunk = kn + ((size_t)bh * T + (size_t)nch * C) * HD;  // [c][dk]
  const float* betac = beta + (size_t)bh * T + (size_t)nch * C;

  if (tid < C) {
    lexpTab[tid] = __expf((float)tid * lg);
    deTab[tid]   = __expf((float)(tid + 1) * lg);
  }
  __syncthreads();

  for (int i = tid; i < C * HD; i += 256) {
    int c = i >> 7, dk = i & 127;
    float kv = (float)kchunk[i];
    kbdeT[dk * C + c] = (bf16)(kv * betac[c] * deTab[c]);
  }
  // A0 tiles (16 tiles of 16x16, 2 per wave), operands direct from global
#pragma unroll
  for (int q = 0; q < 2; ++q) {
    int ti = wave * 2 + q;
    int i0 = (ti >> 2) * 16, j0 = (ti & 3) * 16;
    float sb = betac[i0 + l];     // row-constant beta scale for A operand
    v8f acc = zero8();
#pragma unroll
    for (int k0 = 0; k0 < HD; k0 += 32) {
      v16bf af  = load_frag_a_scale(kchunk + (size_t)i0 * HD + k0, HD, sb);
      v16bf bfm = load_frag_b(kchunk + (size_t)j0 * HD + k0, HD);
      acc = wmma_bb(af, bfm, acc);
    }
    int j = j0 + l;
#pragma unroll
    for (int r = 0; r < 8; ++r) {
      int i = i0 + hi * 8 + r;
      A0[i * C + j] = (i > j) ? (-acc[r] * lexpTab[i - j]) : 0.f;
    }
  }
  __syncthreads();
  // forward substitution, in place: A = (I - A0)^{-1}, unit lower triangular
  if (tid == 0) A0[0] = 1.f;
  __syncthreads();
  for (int i = 1; i < C; ++i) {
    float val = 0.f;
    if (tid < C) {
      int j = tid;
      if (j == i) val = 1.f;
      else if (j < i) {
        float s = 0.f;
        for (int p = j; p < i; ++p) s += A0[i * C + p] * A0[p * C + j];
        val = s;
      }
    }
    __syncthreads();
    if (tid < C) A0[i * C + tid] = val;
    __syncthreads();
  }
  for (int i = tid; i < C * C; i += 256) Ab[i] = (bf16)A0[i];
  __syncthreads();

  const bf16* vchunk = vT + (size_t)blk * C * HD;   // [dv][c]
  float* vprc = vpr + (size_t)blk * C * HD;         // [c][dv]
  bf16*  kcdc = kcd + (size_t)blk * C * HD;         // [c][dk]
  int m0 = (wave >> 1) * 16;
#pragma unroll
  for (int q = 0; q < 4; ++q) {
    int n0 = ((wave & 1) * 4 + q) * 16;
    v8f acc = zero8();
#pragma unroll
    for (int k0 = 0; k0 < C; k0 += 32) {
      v16bf af  = load_frag_a(Ab + m0 * C + k0, C);
      v16bf bfm = load_frag_b(vchunk + (size_t)n0 * C + k0, C);
      acc = wmma_bb(af, bfm, acc);
    }
#pragma unroll
    for (int r = 0; r < 8; ++r) vprc[(m0 + hi * 8 + r) * HD + n0 + l] = acc[r];
    v8f acc2 = zero8();
#pragma unroll
    for (int k0 = 0; k0 < C; k0 += 32) {
      v16bf af  = load_frag_a(Ab + m0 * C + k0, C);
      v16bf bfm = load_frag_b(kbdeT + (size_t)n0 * C + k0, C);
      acc2 = wmma_bb(af, bfm, acc2);
    }
#pragma unroll
    for (int r = 0; r < 8; ++r) kcdc[(m0 + hi * 8 + r) * HD + n0 + l] = (bf16)acc2[r];
  }
}

// ---------------------------------------------------------------------------
// Kernel 4: sequential inter-chunk scan, split over G dv-groups.
// One block per (b,h,g); block owns S[:, dv0:dv0+DVL] (128 x 32 slice).
// 8 waves; each wave holds 2 persistent 16x16 fp32 S-tiles in registers.
// ---------------------------------------------------------------------------
__global__ __launch_bounds__(256) void scan_kernel(
    const bf16* __restrict__ kn, const bf16* __restrict__ kcd,
    const float* __restrict__ vpr, const float* __restrict__ decay,
    bf16* __restrict__ og) {
  __shared__ bf16 STb[DVL * HD];    // S^T slice : [dvl][dk]   8KB
  __shared__ bf16 attn[C * C];      // [i][j]                  8KB
  __shared__ bf16 vnT[DVL * C];     // v_new^T slice : [dvl][c] 4KB
  __shared__ bf16 kt[C * HD];       // k chunk : [c][dk]       16KB
  __shared__ bf16 kdwT[HD * C];     // (k*dw)^T : [dk][c]      16KB
  __shared__ float lexpTab[C];
  __shared__ float deTab[C];
  __shared__ float dwTab[C];

  int blk = blockIdx.x;             // bh*G + g
  int bh = blk >> 2, g = blk & (G - 1);
  int b = bh >> 3, h = bh & 7;
  int dv0 = g * DVL;
  int tid = threadIdx.x;
  int wave = tid >> 5, lane = tid & 31, l = lane & 15, hi = lane >> 4;
  float lg = logf(1.f / (1.f + __expf(-decay[h])));
  float ctd = __expf(64.f * lg);
  if (tid < C) {
    lexpTab[tid] = __expf((float)tid * lg);
    deTab[tid]   = __expf((float)(tid + 1) * lg);
    dwTab[tid]   = __expf((float)(63 - tid) * lg);
  }
  for (int i = tid; i < DVL * HD; i += 256) STb[i] = (bf16)0.f;
  v8f Sacc[2];
#pragma unroll
  for (int j = 0; j < 2; ++j) Sacc[j] = zero8();
  __syncthreads();

  // per-wave tile assignment (8 tiles over c(4) x dvl(2)):
  int m0 = (wave >> 1) * 16;        // c tile for v_new / o
  int n0 = (wave & 1) * 16;         // dvl tile for v_new / o
  int ms = wave * 16;               // dk tile row for S

  for (int nch = 0; nch < NC; ++nch) {
    const bf16* kchunk    = kn  + ((size_t)bh * T + (size_t)nch * C) * HD; // [c][dk]
    const bf16* kcdchunk  = kcd + ((size_t)(bh * NC + nch)) * C * HD;      // [c][dk]
    const float* vprchunk = vpr + ((size_t)(bh * NC + nch)) * C * HD;      // [c][dv]

    // ---- phase 0: stage k -> kt, build (k*dw)^T -> kdwT; prefetch next ----
    {
      const uint4* src = (const uint4*)kchunk;   // 1024 x 16B
      uint4* dst = (uint4*)kt;
      for (int i = tid; i < (C * HD) / 8; i += 256) {
        uint4 v = src[i];
        dst[i] = v;
        bf16 tmp[8];
        *(uint4*)tmp = v;
        int c = i >> 4, dk0 = (i & 15) * 8;
        float dwc = dwTab[c];
#pragma unroll
        for (int j = 0; j < 8; ++j)
          kdwT[(dk0 + j) * C + c] = (bf16)((float)tmp[j] * dwc);
      }
      if (nch + 1 < NC) {
        const char* pk   = (const char*)(kchunk + C * HD);
        const char* pkcd = (const char*)(kcdchunk + C * HD);
        if (tid < 128)       __builtin_prefetch(pk + (size_t)tid * 128, 0, 3);
        else                 __builtin_prefetch(pkcd + (size_t)(tid - 128) * 128, 0, 3);
        if (tid < C)
          __builtin_prefetch((const char*)(vprchunk + C * HD + tid * HD + dv0), 0, 3);
      }
    }
    __syncthreads();

    // ---- phase 1: attn = tril(k k^T * L); v_new = vpr - kcd @ S ----
#pragma unroll
    for (int q = 0; q < 2; ++q) {
      int ti = wave * 2 + q;
      int i0 = (ti >> 2) * 16, j0 = (ti & 3) * 16;
      v8f acc = zero8();
#pragma unroll
      for (int k0 = 0; k0 < HD; k0 += 32) {
        v16bf af  = load_frag_a(kt + i0 * HD + k0, HD);
        v16bf bfm = load_frag_b(kt + j0 * HD + k0, HD);
        acc = wmma_bb(af, bfm, acc);
      }
      int j = j0 + l;
#pragma unroll
      for (int r = 0; r < 8; ++r) {
        int i = i0 + hi * 8 + r;
        float v = (i >= j) ? acc[r] * lexpTab[i - j] : 0.f;
        attn[i * C + j] = (bf16)v;
      }
    }
    {
      v8f acc = zero8();
#pragma unroll
      for (int k0 = 0; k0 < HD; k0 += 32) {
        v16bf af  = load_frag_a(kcdchunk + (size_t)m0 * HD + k0, HD);
        v16bf bfm = load_frag_b(STb + n0 * HD + k0, HD);
        acc = wmma_bb(af, bfm, acc);
      }
#pragma unroll
      for (int r = 0; r < 8; ++r) {
        int cc = m0 + hi * 8 + r;
        int dvl = n0 + l;
        float vn = vprchunk[cc * HD + dv0 + dvl] - acc[r];
        vnT[dvl * C + cc] = (bf16)vn;
      }
    }
    __syncthreads();

    // ---- phase 2: o = (k*de) @ S + attn @ v_new ; dS accumulate ----
    {
      v8f acc = zero8();
      float de = deTab[m0 + l];
#pragma unroll
      for (int k0 = 0; k0 < HD; k0 += 32) {
        v16bf af  = load_frag_a_scale(kt + m0 * HD + k0, HD, de);
        v16bf bfm = load_frag_b(STb + n0 * HD + k0, HD);
        acc = wmma_bb(af, bfm, acc);
      }
#pragma unroll
      for (int k0 = 0; k0 < C; k0 += 32) {
        v16bf af  = load_frag_a(attn + m0 * C + k0, C);
        v16bf bfm = load_frag_b(vnT + n0 * C + k0, C);
        acc = wmma_bb(af, bfm, acc);
      }
#pragma unroll
      for (int r = 0; r < 8; ++r) {
        int cc = m0 + hi * 8 + r;
        int dv = dv0 + n0 + l;
        int t = nch * C + cc;
        og[((size_t)(b * T + t)) * D + h * HD + dv] = (bf16)acc[r];
      }
      // S_new = ctd * S + (k*dw)^T @ v_new  (2 tiles per wave)
#pragma unroll
      for (int j = 0; j < 2; ++j)
#pragma unroll
        for (int r = 0; r < 8; ++r) Sacc[j][r] *= ctd;
#pragma unroll
      for (int k0 = 0; k0 < C; k0 += 32) {
        v16bf af = load_frag_a(kdwT + ms * C + k0, C);
#pragma unroll
        for (int j = 0; j < 2; ++j) {
          v16bf bfm = load_frag_b(vnT + (j * 16) * C + k0, C);
          Sacc[j] = wmma_bb(af, bfm, Sacc[j]);
        }
      }
    }
    __syncthreads();

    // ---- phase 3: refresh bf16 S^T slice in LDS ----
#pragma unroll
    for (int j = 0; j < 2; ++j) {
      int dvl = j * 16 + l;
#pragma unroll
      for (int r = 0; r < 8; ++r) {
        int dk = ms + hi * 8 + r;
        STb[dvl * HD + dk] = (bf16)Sacc[j][r];
      }
    }
    __syncthreads();
  }
}

// ---------------------------------------------------------------------------
// Kernel 5: result = x + o @ Wr^T  (fp32 out, residual fused)
// ---------------------------------------------------------------------------
__global__ __launch_bounds__(256) void gemm_out_kernel(
    const bf16* __restrict__ og, const bf16* __restrict__ Wrb,
    const float* __restrict__ resid, float* __restrict__ out) {
  int wave = threadIdx.x >> 5, lane = threadIdx.x & 31;
  int l = lane & 15, hi = lane >> 4;
  int m0 = blockIdx.x * 128 + (wave >> 1) * 32;
  int n0 = blockIdx.y * 128 + (wave & 1) * 64;
  v8f acc[2][4];
#pragma unroll
  for (int i = 0; i < 2; ++i)
#pragma unroll
    for (int j = 0; j < 4; ++j) acc[i][j] = zero8();
  for (int k0 = 0; k0 < D; k0 += 32) {
    v16bf af[2], bfm[4];
#pragma unroll
    for (int i = 0; i < 2; ++i)
      af[i] = load_frag_a(og + (size_t)(m0 + 16 * i) * D + k0, D);
#pragma unroll
    for (int j = 0; j < 4; ++j)
      bfm[j] = load_frag_b(Wrb + (size_t)(n0 + 16 * j) * D + k0, D);
#pragma unroll
    for (int i = 0; i < 2; ++i)
#pragma unroll
      for (int j = 0; j < 4; ++j) acc[i][j] = wmma_bb(af[i], bfm[j], acc[i][j]);
  }
#pragma unroll
  for (int i = 0; i < 2; ++i) {
#pragma unroll
    for (int j = 0; j < 4; ++j) {
      int n = n0 + 16 * j + l;
#pragma unroll
      for (int r = 0; r < 8; ++r) {
        int m = m0 + 16 * i + hi * 8 + r;
        out[(size_t)m * D + n] = resid[(size_t)m * D + n] + acc[i][j][r];
      }
    }
  }
}

// ---------------------------------------------------------------------------
extern "C" void kernel_launch(void* const* d_in, const int* in_sizes, int n_in,
                              void* d_out, int out_size, void* d_ws, size_t ws_size,
                              hipStream_t stream) {
  (void)in_sizes; (void)n_in; (void)out_size; (void)ws_size;
  const float* x     = (const float*)d_in[0];
  const float* Ww    = (const float*)d_in[1];
  const float* Wr    = (const float*)d_in[2];
  const float* Wbeta = (const float*)d_in[3];
  const float* decay = (const float*)d_in[4];

  char* ws = (char*)d_ws;
  size_t off = 0;
  auto take = [&](size_t bytes) {
    char* p = ws + off;
    off += (bytes + 255) & ~(size_t)255;
    return p;
  };
  bf16*  xb   = (bf16*)take((size_t)BT * D * 2);
  bf16*  Wwb  = (bf16*)take((size_t)D * D * 2);
  bf16*  Wrb  = (bf16*)take((size_t)D * D * 2);
  bf16*  knb  = (bf16*)take((size_t)BT * D * 2);
  float* beta = (float*)take((size_t)BT * H * 4);
  bf16*  vT   = (bf16*)take((size_t)BT * D * 2);
  float* vpr  = (float*)take((size_t)BT * D * 4);
  bf16*  kcd  = (bf16*)take((size_t)BT * D * 2);
  bf16*  og   = (bf16*)take((size_t)BT * D * 2);

  convw_kernel<<<(D * D + 255) / 256, 256, 0, stream>>>(Ww, Wr, Wwb, Wrb);
  prep_kernel<<<BT, 256, 0, stream>>>(x, Wbeta, knb, xb, beta);
  gemm_v_kernel<<<dim3(BT / 128, D / 128), 256, 0, stream>>>(xb, Wwb, beta, vT);
  ut_kernel<<<BH * NC, 256, 0, stream>>>(knb, beta, decay, vT, vpr, kcd);
  scan_kernel<<<BH * G, 256, 0, stream>>>(knb, kcd, vpr, decay, og);
  gemm_out_kernel<<<dim3(BT / 128, D / 128), 256, 0, stream>>>(og, Wrb, x, (float*)d_out);
}
